// SwinTransformerBlock_85341000171896
// MI455X (gfx1250) — matrix operand order repaired and checked by reading, not compiled, not run on hardware
//
#include <hip/hip_runtime.h>
#include <math.h>

// ---------------- types ----------------
typedef __bf16 bf16_t;
typedef __attribute__((ext_vector_type(16))) __bf16 v16bf;
typedef __attribute__((ext_vector_type(8)))  __bf16 v8bf;
typedef __attribute__((ext_vector_type(8)))  float  v8f;
typedef __attribute__((ext_vector_type(4)))  float  v4f;

// ---------------- problem constants ----------------
#define WIN_SZ   7
#define SHIFT_SZ 3
#define NH       16
#define HD       32
#define CDIM     512
#define NTOK     49          // tokens per window
#define HIMG     56
#define BSZ      32
#define NWB      64          // windows per batch image (8x8)
#define NWIN     (BSZ*NWB)   // 2048 windows total
#define MTOT     (NWIN*NTOK) // 100352 tokens (== 6272*16)
#define FFN      2048
#define SCALEQ   0.17677669529663687f   // (1/32)^0.5
#define EPS_LN   1e-5f

// ---------------- helpers ----------------
__device__ __forceinline__ bf16_t f2bf(float f) {
  union { float f; unsigned u; } in; in.f = f;
  unsigned r = (in.u + 0x7FFFu + ((in.u >> 16) & 1u)) >> 16;   // RNE
  union { unsigned short s; bf16_t b; } out; out.s = (unsigned short)r;
  return out.b;
}

__device__ __forceinline__ v16bf make_frag(v8bf lo, v8bf hi) {
  v16bf r;
#pragma unroll
  for (int i = 0; i < 8; ++i) { r[i] = lo[i]; r[i + 8] = hi[i]; }
  return r;
}

__device__ __forceinline__ v16bf zero_frag() {
  bf16_t z = f2bf(0.0f);
  v16bf r;
#pragma unroll
  for (int i = 0; i < 16; ++i) r[i] = z;
  return r;
}

__device__ __forceinline__ float wave_sum32(float s) {
#pragma unroll
  for (int off = 16; off >= 1; off >>= 1) s += __shfl_xor(s, off, 32);
  return s;
}

// ---------------- weight repack: fp32 (Nout,K) row-major -> bf16 K-panel layout
// Wp[((k/32)*Nout + n)*32 + (k%32)] = w[n*K + k]   (B[k][n] = w[n][k])
__global__ __launch_bounds__(256)
void repack_kernel(const float* __restrict__ w, bf16_t* __restrict__ wp,
                   int Nout, int K) {
  size_t i = (size_t)blockIdx.x * 256 + threadIdx.x;
  if (i >= (size_t)Nout * K) return;
  int n = (int)(i / K);
  int k = (int)(i % K);
  wp[((size_t)(k >> 5) * Nout + n) * 32 + (k & 31)] = f2bf(w[(size_t)n * K + k]);
}

// ---------------- LayerNorm (one wave per token, 16 ch/lane)
// shifted=1: write to shifted-window token order; shifted=0: identity order
__global__ __launch_bounds__(256)
void ln_kernel(const float* __restrict__ src, const float* __restrict__ g,
               const float* __restrict__ b, bf16_t* __restrict__ dst, int shifted) {
  int lane = threadIdx.x & 31;
  int wv   = threadIdx.x >> 5;
  int token = blockIdx.x * 8 + wv;
  const float* p = src + (size_t)token * CDIM + lane * 16;

  float v[16];
#pragma unroll
  for (int i = 0; i < 4; ++i) {
    v4f t = *(const v4f*)(p + i * 4);
#pragma unroll
    for (int j = 0; j < 4; ++j) v[i * 4 + j] = t[j];
  }
  float s = 0.f;
#pragma unroll
  for (int i = 0; i < 16; ++i) s += v[i];
  s = wave_sum32(s);
  float mu = s * (1.0f / 512.0f);
  float q = 0.f;
#pragma unroll
  for (int i = 0; i < 16; ++i) { float d = v[i] - mu; q += d * d; }
  q = wave_sum32(q);
  float inv = rsqrtf(q * (1.0f / 512.0f) + EPS_LN);

  size_t dtok;
  if (shifted) {
    int b_ = token / (HIMG * HIMG);
    int ij = token % (HIMG * HIMG);
    int i0 = ij / HIMG, j0 = ij % HIMG;
    int i = i0 - SHIFT_SZ; if (i < 0) i += HIMG;
    int j = j0 - SHIFT_SZ; if (j < 0) j += HIMG;
    int win = b_ * NWB + (i / WIN_SZ) * 8 + (j / WIN_SZ);
    int n   = (i % WIN_SZ) * WIN_SZ + (j % WIN_SZ);
    dtok = (size_t)win * NTOK + n;
  } else {
    dtok = (size_t)token;
  }

  v8bf o0, o1;
#pragma unroll
  for (int e = 0; e < 8; ++e) {
    int c0 = lane * 16 + e, c1 = lane * 16 + 8 + e;
    o0[e] = f2bf((v[e]     - mu) * inv * g[c0] + b[c0]);
    o1[e] = f2bf((v[e + 8] - mu) * inv * g[c1] + b[c1]);
  }
  bf16_t* d = dst + dtok * CDIM + lane * 16;
  *(v8bf*)d       = o0;
  *(v8bf*)(d + 8) = o1;
}

// ---------------- generic WMMA GEMM: C = A(bf16 MxK) * Wp(bf16 KxN panels) + bias
// block = 256 threads = 8 waves (2 M-tiles x 4 wave-cols), block tile 32x256.
// MODE 0: qkv split  -> b0=q(*SCALEQ) b1=k b2=v, head-major (win,head,n,d)
// MODE 1: proj       -> fout[orig] = resid[orig] + acc + bias (window reverse + unshift)
// MODE 2: fc1 + GELU -> b0[row*2048+col]
// MODE 3: fc2        -> fout[row*512+col] += acc + bias (in-place residual)
template <int MODE>
__global__ __launch_bounds__(256)
void gemm_kernel(const bf16_t* __restrict__ A, const bf16_t* __restrict__ Wp,
                 const float* __restrict__ bias, int K, int Nn,
                 float* __restrict__ fout, const float* __restrict__ resid,
                 bf16_t* __restrict__ b0, bf16_t* __restrict__ b1,
                 bf16_t* __restrict__ b2) {
  int lane = threadIdx.x & 31;
  int wave = threadIdx.x >> 5;
  int wm = wave >> 2, wn = wave & 3;
  int rowbase = blockIdx.x * 32 + wm * 16;
  int colbase = blockIdx.y * 256 + wn * 64;
  int lr = lane & 15, kh = lane >> 4;

  v8f acc[4];
#pragma unroll
  for (int ct = 0; ct < 4; ++ct)
#pragma unroll
    for (int i = 0; i < 8; ++i) acc[ct][i] = 0.0f;

  for (int kb = 0; kb < K; kb += 32) {
    const bf16_t* ap = A + (size_t)(rowbase + lr) * K + kb + kh * 16;
    v8bf a0 = *(const v8bf*)ap;
    v8bf a1 = *(const v8bf*)(ap + 8);
    v16bf af = make_frag(a0, a1);
#pragma unroll
    for (int ct = 0; ct < 4; ++ct) {
      const bf16_t* bp =
          Wp + ((size_t)(kb >> 5) * Nn + colbase + ct * 16 + lr) * 32 + kh * 16;
      v16bf bfrag = *(const v16bf*)bp;
      acc[ct] = __builtin_amdgcn_wmma_f32_16x16x32_bf16(
          false, af, false, bfrag, (short)0, acc[ct], false, false);
    }
  }

#pragma unroll
  for (int ct = 0; ct < 4; ++ct) {
#pragma unroll
    for (int r = 0; r < 8; ++r) {
      int row = rowbase + r + (kh ? 8 : 0);
      int col = colbase + ct * 16 + lr;
      float val = acc[ct][r] + bias[col];
      if (MODE == 0) {
        int part = col >> 9;               // 0:q 1:k 2:v
        int hc   = col & 511;
        int head = hc >> 5, d = hc & 31;
        int win = row / NTOK, n = row % NTOK;
        size_t dst = (((size_t)win * NH + head) * NTOK + n) * HD + d;
        if (part == 0)      b0[dst] = f2bf(val * SCALEQ);
        else if (part == 1) b1[dst] = f2bf(val);
        else                b2[dst] = f2bf(val);
      } else if (MODE == 1) {
        int win = row / NTOK, n = row % NTOK;
        int b_ = win >> 6, wib = win & 63;
        int i = (wib >> 3) * WIN_SZ + n / WIN_SZ;
        int j = (wib & 7)  * WIN_SZ + n % WIN_SZ;
        int i0 = i + SHIFT_SZ; if (i0 >= HIMG) i0 -= HIMG;
        int j0 = j + SHIFT_SZ; if (j0 >= HIMG) j0 -= HIMG;
        size_t tok = (size_t)b_ * (HIMG * HIMG) + i0 * HIMG + j0;
        fout[tok * CDIM + col] = resid[tok * CDIM + col] + val;
      } else if (MODE == 2) {
        float gel = 0.5f * val * (1.0f + erff(val * 0.70710678118654752f));
        b0[(size_t)row * FFN + col] = f2bf(gel);
      } else {
        fout[(size_t)row * CDIM + col] += val;
      }
    }
  }
}

// ---------------- attention: one block (128 thr = 4 waves) per (window, head)
__global__ __launch_bounds__(128)
void attn_kernel(const bf16_t* __restrict__ q, const bf16_t* __restrict__ k,
                 const bf16_t* __restrict__ v, const float* __restrict__ rpb,
                 const float* __restrict__ mask, bf16_t* __restrict__ o) {
  __shared__ float Ssm[64 * 64];
  __shared__ __attribute__((aligned(32))) bf16_t Pb[64 * 64];
  __shared__ __attribute__((aligned(32))) bf16_t Vt[32 * 64];

  int tid  = threadIdx.x;
  int lane = tid & 31, wave = tid >> 5;
  int lr = lane & 15, kh = lane >> 4;
  int win = blockIdx.x >> 4, head = blockIdx.x & 15;

  const bf16_t* qb = q + (((size_t)win * NH + head) * NTOK) * HD;
  const bf16_t* kb = k + (((size_t)win * NH + head) * NTOK) * HD;
  const bf16_t* vb = v + (((size_t)win * NH + head) * NTOK) * HD;

  // stage V transposed (32 x 64), zero-pad cols 49..63
  bf16_t bz = f2bf(0.0f);
  for (int idx = tid; idx < 32 * 64; idx += 128) {
    int d = idx >> 6, m = idx & 63;
    Vt[idx] = (m < NTOK) ? vb[m * HD + d] : bz;
  }

  // ---- S = Q * K^T  (each wave: rows [16*wave,16*wave+16), 4 col tiles, K=32 = 1 wmma)
  int qrow = wave * 16 + lr;
  v16bf af = zero_frag();
  if (qrow < NTOK) {
    const bf16_t* qp = qb + qrow * HD + kh * 16;
    af = make_frag(*(const v8bf*)qp, *(const v8bf*)(qp + 8));
  }
  v8f sacc[4];
#pragma unroll
  for (int ct = 0; ct < 4; ++ct)
#pragma unroll
    for (int i = 0; i < 8; ++i) sacc[ct][i] = 0.0f;

#pragma unroll
  for (int ct = 0; ct < 4; ++ct) {
    int colm = ct * 16 + lr;
    v16bf bfrag = zero_frag();
    if (colm < NTOK) bfrag = *(const v16bf*)(kb + colm * HD + kh * 16);
    sacc[ct] = __builtin_amdgcn_wmma_f32_16x16x32_bf16(
        false, af, false, bfrag, (short)0, sacc[ct], false, false);
  }
#pragma unroll
  for (int ct = 0; ct < 4; ++ct)
#pragma unroll
    for (int r = 0; r < 8; ++r) {
      int row = wave * 16 + r + (kh ? 8 : 0);
      int col = ct * 16 + lr;
      Ssm[row * 64 + col] = sacc[ct][r];
    }
  __syncthreads();

  // ---- softmax per row (bias + mask fused), write bf16 P, zero padding
  if (tid < 64) {
    int r = tid;
    if (r < NTOK) {
      int ni = r / WIN_SZ, nj = r % WIN_SZ;
      const float* mrow = mask + ((size_t)(win & 63) * NTOK + r) * NTOK;
      float mx = -1e30f;
      for (int m = 0; m < NTOK; ++m) {
        int mi = m / WIN_SZ, mj = m % WIN_SZ;
        int rel = (ni - mi + WIN_SZ - 1) * (2 * WIN_SZ - 1) + (nj - mj + WIN_SZ - 1);
        float s = Ssm[r * 64 + m] + rpb[rel * NH + head] + mrow[m];
        Ssm[r * 64 + m] = s;
        mx = fmaxf(mx, s);
      }
      float sum = 0.f;
      for (int m = 0; m < NTOK; ++m) {
        float e = expf(Ssm[r * 64 + m] - mx);
        Ssm[r * 64 + m] = e;
        sum += e;
      }
      float inv = 1.0f / sum;
      for (int m = 0; m < NTOK; ++m) Pb[r * 64 + m] = f2bf(Ssm[r * 64 + m] * inv);
      for (int m = NTOK; m < 64; ++m) Pb[r * 64 + m] = bz;
    } else {
      for (int m = 0; m < 64; ++m) Pb[r * 64 + m] = bz;
    }
  }
  __syncthreads();

  // ---- O = P * V  (K padded to 64 -> 2 k-steps; N = 32 -> 2 tiles)
  v8f oacc[2];
#pragma unroll
  for (int ct = 0; ct < 2; ++ct)
#pragma unroll
    for (int i = 0; i < 8; ++i) oacc[ct][i] = 0.0f;

#pragma unroll
  for (int ks = 0; ks < 2; ++ks) {
    const bf16_t* pp = &Pb[(wave * 16 + lr) * 64 + ks * 32 + kh * 16];
    v16bf pf = make_frag(*(const v8bf*)pp, *(const v8bf*)(pp + 8));
#pragma unroll
    for (int ct = 0; ct < 2; ++ct) {
      const bf16_t* vp = &Vt[(ct * 16 + lr) * 64 + ks * 32 + kh * 16];
      v16bf vf = *(const v16bf*)vp;
      oacc[ct] = __builtin_amdgcn_wmma_f32_16x16x32_bf16(
          false, pf, false, vf, (short)0, oacc[ct], false, false);
    }
  }
#pragma unroll
  for (int ct = 0; ct < 2; ++ct)
#pragma unroll
    for (int r = 0; r < 8; ++r) {
      int row = wave * 16 + r + (kh ? 8 : 0);
      int col = ct * 16 + lr;
      if (row < NTOK)
        o[((size_t)win * NTOK + row) * CDIM + head * HD + col] = f2bf(oacc[ct][r]);
    }
}

// ---------------- launch ----------------
extern "C" void kernel_launch(void* const* d_in, const int* in_sizes, int n_in,
                              void* d_out, int out_size, void* d_ws, size_t ws_size,
                              hipStream_t stream) {
  (void)in_sizes; (void)n_in; (void)out_size; (void)ws_size;
  const float* x      = (const float*)d_in[0];
  const float* mask   = (const float*)d_in[1];
  const float* ln1_g  = (const float*)d_in[2];
  const float* ln1_b  = (const float*)d_in[3];
  const float* qkv_w  = (const float*)d_in[4];
  const float* qkv_b  = (const float*)d_in[5];
  const float* proj_w = (const float*)d_in[6];
  const float* proj_b = (const float*)d_in[7];
  const float* rpb    = (const float*)d_in[8];
  const float* ln2_g  = (const float*)d_in[9];
  const float* ln2_b  = (const float*)d_in[10];
  const float* fc1_w  = (const float*)d_in[11];
  const float* fc1_b  = (const float*)d_in[12];
  const float* fc2_w  = (const float*)d_in[13];
  const float* fc2_b  = (const float*)d_in[14];
  float* out = (float*)d_out;

  const size_t X = (size_t)MTOT * CDIM * sizeof(bf16_t);  // 102,760,448 B
  char* ws = (char*)d_ws;
  bf16_t* xw   = (bf16_t*)(ws);           // LN output (win order, then natural order)
  bf16_t* qbuf = (bf16_t*)(ws + 1 * X);
  bf16_t* kbuf = (bf16_t*)(ws + 2 * X);
  bf16_t* vbuf = (bf16_t*)(ws + 3 * X);
  bf16_t* obuf = (bf16_t*)(ws + 4 * X);
  bf16_t* h1   = (bf16_t*)(ws + 1 * X);   // aliases q/k/v/o (dead by fc1), 4X bytes
  char* wbase = ws + 5 * X;
  bf16_t* wq = (bf16_t*)(wbase);                                  // 1536x512
  bf16_t* wp = (bf16_t*)(wbase + (size_t)1536 * 512 * 2);         // 512x512
  bf16_t* w1 = (bf16_t*)(wbase + (size_t)(1536 + 512) * 512 * 2); // 2048x512
  bf16_t* w2 = (bf16_t*)(wbase + (size_t)(1536 + 512 + 2048) * 512 * 2); // 2048x512

  // weight repacks (fp32 -> bf16 K-panel layout)
  repack_kernel<<<(1536 * 512 + 255) / 256, 256, 0, stream>>>(qkv_w, wq, 1536, 512);
  repack_kernel<<<(512 * 512 + 255) / 256, 256, 0, stream>>>(proj_w, wp, 512, 512);
  repack_kernel<<<(2048 * 512 + 255) / 256, 256, 0, stream>>>(fc1_w, w1, 2048, 512);
  repack_kernel<<<(2048 * 512 + 255) / 256, 256, 0, stream>>>(fc2_w, w2, 512, 2048);

  // LN1 + shift + window partition
  ln_kernel<<<MTOT / 8, 256, 0, stream>>>(x, ln1_g, ln1_b, xw, 1);

  // QKV
  gemm_kernel<0><<<dim3(MTOT / 32, 1536 / 256), 256, 0, stream>>>(
      xw, wq, qkv_b, 512, 1536, nullptr, nullptr, qbuf, kbuf, vbuf);

  // attention
  attn_kernel<<<NWIN * NH, 128, 0, stream>>>(qbuf, kbuf, vbuf, rpb, mask, obuf);

  // proj + window-reverse + unshift + residual -> out
  gemm_kernel<1><<<dim3(MTOT / 32, 512 / 256), 256, 0, stream>>>(
      obuf, wp, proj_b, 512, 512, out, x, nullptr, nullptr, nullptr);

  // LN2 (natural order)
  ln_kernel<<<MTOT / 8, 256, 0, stream>>>(out, ln2_g, ln2_b, xw, 0);

  // fc1 + GELU
  gemm_kernel<2><<<dim3(MTOT / 32, 2048 / 256), 256, 0, stream>>>(
      xw, w1, fc1_b, 512, 2048, nullptr, nullptr, h1, nullptr, nullptr);

  // fc2 + residual (in place on out)
  gemm_kernel<3><<<dim3(MTOT / 32, 512 / 256), 256, 0, stream>>>(
      h1, w2, fc2_b, 2048, 512, out, nullptr, nullptr, nullptr, nullptr);
}
